// GNNPolicy_ancon_37838661878453
// MI455X (gfx1250) — matrix operand order repaired and checked by reading, not compiled, not run on hardware
//
#include <hip/hip_runtime.h>
#include <hip/hip_bf16.h>
#include <math.h>

#define EN 64
#define NC 8
#define NNODES 524288
#define TPW 8
#define BLOCKS_P (NNODES / (16 * 8 * TPW))   // 512

typedef _Float16 h16;
typedef __attribute__((ext_vector_type(16))) _Float16 v16h;
typedef __attribute__((ext_vector_type(8)))  float    v8f;

union FragU { float4 f4[2]; v16h h; };

// ---- monotone float<->u32 key for atomic segment-max ----
__device__ inline unsigned f2key(float f) {
    unsigned b = __float_as_uint(f);
    return (b & 0x80000000u) ? ~b : (b | 0x80000000u);
}
__device__ inline float key2f(unsigned k) {
    unsigned b = (k & 0x80000000u) ? (k & 0x7fffffffu) : ~k;
    return __uint_as_float(b);
}
#define NEG_INF_KEY 0x007FFFFFu   // f2key(-inf)

__device__ inline float red16(float v) {
    v += __shfl_xor(v, 1, 16);
    v += __shfl_xor(v, 2, 16);
    v += __shfl_xor(v, 4, 16);
    v += __shfl_xor(v, 8, 16);
    return v;
}

__device__ inline v8f wmma_f16(v16h a, v16h b, v8f c) {
    return __builtin_amdgcn_wmma_f32_16x16x32_f16(
        false, a, false, b, (short)0, c, false, false);
}

// A fragment (16x32 f16, ISA 7.12.2): lane l holds row M=l&15;
// elems 0..7  -> K = koff+e      (koff = 8*(l>>4))
// elems 8..15 -> K = 16+koff+(e-8)
__device__ inline v16h load_a_global(const float* __restrict__ xb, int lane, int kc) {
    int M = lane & 15, koff = (lane >> 4) * 8;
    const float* r = xb + M * EN + kc * 32;
    float4 f0 = *(const float4*)(r + koff);
    float4 f1 = *(const float4*)(r + koff + 4);
    float4 f2 = *(const float4*)(r + 16 + koff);
    float4 f3 = *(const float4*)(r + 16 + koff + 4);
    v16h a;
    a[0]=(h16)f0.x;  a[1]=(h16)f0.y;  a[2]=(h16)f0.z;  a[3]=(h16)f0.w;
    a[4]=(h16)f1.x;  a[5]=(h16)f1.y;  a[6]=(h16)f1.z;  a[7]=(h16)f1.w;
    a[8]=(h16)f2.x;  a[9]=(h16)f2.y;  a[10]=(h16)f2.z; a[11]=(h16)f2.w;
    a[12]=(h16)f3.x; a[13]=(h16)f3.y; a[14]=(h16)f3.z; a[15]=(h16)f3.w;
    return a;
}

__device__ inline v16h load_a_lds(const h16* __restrict__ xs, int lane, int kc) {
    int M = lane & 15, koff = (lane >> 4) * 8;
    const h16* r = xs + M * EN + kc * 32;
    FragU f;
    f.f4[0] = *(const float4*)(r + koff);
    f.f4[1] = *(const float4*)(r + 16 + koff);
    return f.h;
}

// B fragment (32x16 f16): lane l holds col N = (l&15)+16*ntile,
// elems 0..7 -> K = 8*(l>>4) + e, elems 8..15 -> K = 16 + 8*(l>>4) + (e-8).  W is row-major [n][k] f16 in LDS.
__device__ inline v16h load_b_lds(const h16* __restrict__ W, int lane, int ntile, int kc) {
    int n = (lane & 15) + ntile * 16;
    int koff = (lane >> 4) * 8;
    const h16* r = W + n * EN + kc * 32 + koff;
    FragU f;
    f.f4[0] = *(const float4*)(r);
    f.f4[1] = *(const float4*)(r + 16);
    return f.h;
}

// ---------------- P0: prep (q, u-table, init accumulators) ----------------
__global__ __launch_bounds__(256) void k_prep(
    const float* __restrict__ sem, const float* __restrict__ Wqkv,
    const float* __restrict__ bqkv,
    float* __restrict__ u, unsigned* __restrict__ mkeys,
    float* __restrict__ attnAcc, float* __restrict__ oldAcc,
    float* __restrict__ Zacc, float* __restrict__ cntAcc)
{
    __shared__ float qL[NC * EN];
    int tid = threadIdx.x;
    for (int i = tid; i < NC * EN; i += 256) {
        int c = i >> 6, o = i & 63;
        float s = bqkv[o];
        for (int j = 0; j < EN; ++j) s += sem[c * EN + j] * Wqkv[o * EN + j];
        qL[i] = s * 0.25f;   // 1/sqrt(DH=16)
    }
    __syncthreads();
    // u[c][j][h] = sum_d q[c,16h+d] * Wk[16h+d, j]   (Wk = Wqkv rows 64..127)
    for (int i = tid; i < NC * EN * 4; i += 256) {
        int c = i >> 8, j = (i >> 2) & 63, h = i & 3;
        float s = 0.f;
        for (int d = 0; d < 16; ++d)
            s += qL[c * EN + 16 * h + d] * Wqkv[(EN + 16 * h + d) * EN + j];
        u[i] = s;
    }
    for (int i = tid; i < NC * EN; i += 256) { attnAcc[i] = 0.f; oldAcc[i] = 0.f; }
    if (tid < NC * 4) { mkeys[tid] = NEG_INF_KEY; Zacc[tid] = 0.f; }
    if (tid < NC) cntAcc[tid] = 0.f;
}

// ---------------- P1: xs GEMM + scores + segment max ----------------
__global__ __launch_bounds__(256) void k_pass1(
    const float* __restrict__ x, const int* __restrict__ cls,
    const float* __restrict__ Ws, const float* __restrict__ bs,
    const float* __restrict__ u, unsigned* __restrict__ mkeys)
{
    __shared__ __align__(16) h16   WL[EN * EN];
    __shared__ __align__(16) float uL[NC * EN * 4];
    __shared__ unsigned mkL[NC * 4];
    int tid = threadIdx.x;
    for (int i = tid; i < EN * EN; i += 256) WL[i] = (h16)Ws[i];
    for (int i = tid; i < NC * EN * 4; i += 256) uL[i] = u[i];
    if (tid < NC * 4) mkL[tid] = NEG_INF_KEY;
    __syncthreads();

    int lane = tid & 31, wave = tid >> 5, l15 = lane & 15, half = lane >> 4;
    float bsv[4];
#pragma unroll
    for (int t = 0; t < 4; ++t) bsv[t] = bs[t * 16 + l15];

    for (int it = 0; it < TPW; ++it) {
        int tile = (blockIdx.x * 8 + wave) * TPW + it;
        int rowBase = tile * 16;
        const float* xb = x + (size_t)rowBase * EN;
        v16h a0 = load_a_global(xb, lane, 0);
        v16h a1 = load_a_global(xb, lane, 1);
        v8f acc[4];
#pragma unroll
        for (int t = 0; t < 4; ++t) {
            v8f z = {0.f,0.f,0.f,0.f,0.f,0.f,0.f,0.f};
            z = wmma_f16(a0, load_b_lds(WL, lane, t, 0), z);
            z = wmma_f16(a1, load_b_lds(WL, lane, t, 1), z);
#pragma unroll
            for (int r = 0; r < 8; ++r) z[r] += bsv[t];
            acc[t] = z;
        }
#pragma unroll
        for (int r = 0; r < 8; ++r) {
            int M = r + half * 8;
            int c = cls[rowBase + M];
            float s0 = 0.f, s1 = 0.f, s2 = 0.f, s3 = 0.f;
#pragma unroll
            for (int t = 0; t < 4; ++t) {
                float xv = acc[t][r];
                float4 uv = *(const float4*)&uL[(c * EN + t * 16 + l15) * 4];
                s0 += xv * uv.x; s1 += xv * uv.y; s2 += xv * uv.z; s3 += xv * uv.w;
            }
            s0 = red16(s0); s1 = red16(s1); s2 = red16(s2); s3 = red16(s3);
            if (l15 < 4) {
                float sv = (l15 == 0) ? s0 : (l15 == 1) ? s1 : (l15 == 2) ? s2 : s3;
                atomicMax(&mkL[c * 4 + l15], f2key(sv));
            }
        }
    }
    __syncthreads();
    if (tid < NC * 4) atomicMax(&mkeys[tid], mkL[tid]);
}

// ------- P2: recompute xs, softmax numerators, v GEMM, segment sums -------
__global__ __launch_bounds__(256) void k_pass2(
    const float* __restrict__ x, const int* __restrict__ cls,
    const float* __restrict__ Ws, const float* __restrict__ bs,
    const float* __restrict__ Wv, const float* __restrict__ bv,
    const float* __restrict__ u, const unsigned* __restrict__ mkeys,
    float* __restrict__ attnAcc, float* __restrict__ oldAcc,
    float* __restrict__ Zacc, float* __restrict__ cntAcc)
{
    __shared__ __align__(16) h16   WsL[EN * EN];
    __shared__ __align__(16) h16   WvL[EN * EN];
    __shared__ __align__(16) float uL[NC * EN * 4];
    __shared__ float mval[NC * 4];
    __shared__ float attnL[NC * EN], oldL[NC * EN], ZL[NC * 4], cntL[NC];
    __shared__ __align__(16) h16   xsW[8][16 * EN];
    __shared__ float pW[8][16 * 4];

    int tid = threadIdx.x;
    for (int i = tid; i < EN * EN; i += 256) { WsL[i] = (h16)Ws[i]; WvL[i] = (h16)Wv[i]; }
    for (int i = tid; i < NC * EN * 4; i += 256) uL[i] = u[i];
    if (tid < NC * 4) { mval[tid] = key2f(mkeys[tid]); ZL[tid] = 0.f; }
    for (int i = tid; i < NC * EN; i += 256) { attnL[i] = 0.f; oldL[i] = 0.f; }
    if (tid < NC) cntL[tid] = 0.f;
    __syncthreads();

    int lane = tid & 31, wave = tid >> 5, l15 = lane & 15, half = lane >> 4;
    float bsv[4], bvv[4];
#pragma unroll
    for (int t = 0; t < 4; ++t) { bsv[t] = bs[t * 16 + l15]; bvv[t] = bv[t * 16 + l15]; }

    for (int it = 0; it < TPW; ++it) {
        int tile = (blockIdx.x * 8 + wave) * TPW + it;
        int rowBase = tile * 16;
        const float* xb = x + (size_t)rowBase * EN;
        v16h a0 = load_a_global(xb, lane, 0);
        v16h a1 = load_a_global(xb, lane, 1);
        v8f xsa[4];
#pragma unroll
        for (int t = 0; t < 4; ++t) {
            v8f z = {0.f,0.f,0.f,0.f,0.f,0.f,0.f,0.f};
            z = wmma_f16(a0, load_b_lds(WsL, lane, t, 0), z);
            z = wmma_f16(a1, load_b_lds(WsL, lane, t, 1), z);
#pragma unroll
            for (int r = 0; r < 8; ++r) z[r] += bsv[t];
            xsa[t] = z;
        }
        // scores -> p, Z, stage xs (f16) and old-sums
#pragma unroll
        for (int r = 0; r < 8; ++r) {
            int M = r + half * 8;
            int c = cls[rowBase + M];
            float s0 = 0.f, s1 = 0.f, s2 = 0.f, s3 = 0.f;
#pragma unroll
            for (int t = 0; t < 4; ++t) {
                float xv = xsa[t][r];
                float4 uv = *(const float4*)&uL[(c * EN + t * 16 + l15) * 4];
                s0 += xv * uv.x; s1 += xv * uv.y; s2 += xv * uv.z; s3 += xv * uv.w;
            }
            s0 = red16(s0); s1 = red16(s1); s2 = red16(s2); s3 = red16(s3);
            if (l15 < 4) {
                float sv = (l15 == 0) ? s0 : (l15 == 1) ? s1 : (l15 == 2) ? s2 : s3;
                float pv = __expf(sv - mval[c * 4 + l15]);
                atomicAdd(&ZL[c * 4 + l15], pv);
                pW[wave][M * 4 + l15] = pv;
            }
#pragma unroll
            for (int t = 0; t < 4; ++t) {
                xsW[wave][M * EN + t * 16 + l15] = (h16)xsa[t][r];
                atomicAdd(&oldL[c * EN + t * 16 + l15], xsa[t][r]);
            }
        }
        if (lane < 16) atomicAdd(&cntL[cls[rowBase + lane]], 1.f);
        __syncthreads();

        // vv = xs @ Wv^T + bv, accumulate p * vv per class
        v16h xa0 = load_a_lds(xsW[wave], lane, 0);
        v16h xa1 = load_a_lds(xsW[wave], lane, 1);
        v8f va[4];
#pragma unroll
        for (int t = 0; t < 4; ++t) {
            v8f z = {0.f,0.f,0.f,0.f,0.f,0.f,0.f,0.f};
            z = wmma_f16(xa0, load_b_lds(WvL, lane, t, 0), z);
            z = wmma_f16(xa1, load_b_lds(WvL, lane, t, 1), z);
#pragma unroll
            for (int r = 0; r < 8; ++r) z[r] += bvv[t];
            va[t] = z;
        }
#pragma unroll
        for (int r = 0; r < 8; ++r) {
            int M = r + half * 8;
            int c = cls[rowBase + M];
#pragma unroll
            for (int t = 0; t < 4; ++t) {
                float pv = pW[wave][M * 4 + t];
                atomicAdd(&attnL[c * EN + t * 16 + l15], pv * va[t][r]);
            }
        }
        __syncthreads();
    }

    for (int i = tid; i < NC * EN; i += 256) {
        atomicAdd(&attnAcc[i], attnL[i]);
        atomicAdd(&oldAcc[i],  oldL[i]);
    }
    if (tid < NC * 4) atomicAdd(&Zacc[tid], ZL[tid]);
    if (tid < NC)     atomicAdd(&cntAcc[tid], cntL[tid]);
}

// ---------------- P3: finalize (tiny) ----------------
__global__ __launch_bounds__(256) void k_final(
    const float* __restrict__ sem,
    const float* __restrict__ Wo, const float* __restrict__ bo,
    const float* __restrict__ Wr, const float* __restrict__ br,
    const float* __restrict__ Wg, const float* __restrict__ bg,
    const float* __restrict__ gamma, const float* __restrict__ beta,
    const float* __restrict__ attnAcc, const float* __restrict__ oldAcc,
    const float* __restrict__ Zacc, const float* __restrict__ cntAcc,
    float* __restrict__ fused)
{
    __shared__ float attn[NC*EN], oldv[NC*EN], attno[NC*EN], newv[NC*EN], yv[NC*EN];
    __shared__ float mu[NC], rstd[NC];
    int tid = threadIdx.x;
    for (int i = tid; i < NC * EN; i += 256) {
        int c = i >> 6, j = i & 63;
        attn[i] = attnAcc[i] / Zacc[c * 4 + (j >> 4)];
        oldv[i] = oldAcc[i] / fmaxf(cntAcc[c], 1.f);
    }
    __syncthreads();
    for (int i = tid; i < NC * EN; i += 256) {
        int c = i >> 6, n = i & 63;
        float s = bo[n];
        for (int j = 0; j < EN; ++j) s += attn[c * EN + j] * Wo[n * EN + j];
        attno[i] = s;
    }
    __syncthreads();
    for (int i = tid; i < NC * EN; i += 256) {
        int c = i >> 6, n = i & 63;
        float s = br[n];
        for (int j = 0; j < EN; ++j) s += sem[c * EN + j]   * Wr[n * 128 + j];
        for (int j = 0; j < EN; ++j) s += attno[c * EN + j] * Wr[n * 128 + 64 + j];
        newv[i] = s;
    }
    __syncthreads();
    for (int i = tid; i < NC * EN; i += 256) {
        int c = i >> 6, n = i & 63;
        float s = bg[n];
        for (int j = 0; j < EN; ++j) s += oldv[c * EN + j] * Wg[n * 128 + j];
        for (int j = 0; j < EN; ++j) s += newv[c * EN + j] * Wg[n * 128 + 64 + j];
        float g = 1.f / (1.f + __expf(-s));
        yv[i] = g * oldv[i] + (1.f - g) * newv[i];
    }
    __syncthreads();
    if (tid < NC) {
        float m = 0.f;
        for (int j = 0; j < EN; ++j) m += yv[tid * EN + j];
        m *= (1.f / 64.f);
        float var = 0.f;
        for (int j = 0; j < EN; ++j) { float d = yv[tid * EN + j] - m; var += d * d; }
        var *= (1.f / 64.f);
        mu[tid] = m; rstd[tid] = rsqrtf(var + 1e-5f);
    }
    __syncthreads();
    for (int i = tid; i < NC * EN; i += 256) {
        int c = i >> 6, n = i & 63;
        fused[i] = (yv[i] - mu[c]) * rstd[c] * gamma[n] + beta[n];
    }
}

// ---------------- P4: out = fused[cls] * x ----------------
__global__ __launch_bounds__(256) void k_scatter(
    const float* __restrict__ x, const int* __restrict__ cls,
    const float* __restrict__ fused, float* __restrict__ out)
{
    __shared__ float fL[NC * EN];
    int tid = threadIdx.x;
    for (int i = tid; i < NC * EN; i += 256) fL[i] = fused[i];
    __syncthreads();
    size_t idx = (size_t)blockIdx.x * 256 + tid;   // float4 index
    size_t row = idx >> 4;
    int q = (int)(idx & 15);
    int c = cls[row];
    float4 xv = ((const float4*)x)[idx];
    const float* f = &fL[c * EN + q * 4];
    float4 o;
    o.x = xv.x * f[0]; o.y = xv.y * f[1]; o.z = xv.z * f[2]; o.w = xv.w * f[3];
    ((float4*)out)[idx] = o;
}

// ---------------- host side ----------------
static void run_side(const float* x, const float* sem, const int* cls,
                     const float* Ws, const float* bs,
                     const float* Wqkv, const float* bqkv,
                     const float* Wo, const float* bo,
                     const float* Wr, const float* br,
                     const float* Wg, const float* bg,
                     const float* gamma, const float* beta,
                     float* out, float* wsf, hipStream_t stream)
{
    float*    u       = wsf;                   // 2048
    float*    attnAcc = wsf + 2048;            // 512
    float*    oldAcc  = wsf + 2560;            // 512
    float*    Zacc    = wsf + 3072;            // 32
    float*    cntAcc  = wsf + 3104;            // 8
    float*    fused   = wsf + 3136;            // 512
    unsigned* mkeys   = (unsigned*)(wsf + 3648); // 32

    k_prep<<<1, 256, 0, stream>>>(sem, Wqkv, bqkv, u, mkeys, attnAcc, oldAcc, Zacc, cntAcc);
    k_pass1<<<BLOCKS_P, 256, 0, stream>>>(x, cls, Ws, bs, u, mkeys);
    k_pass2<<<BLOCKS_P, 256, 0, stream>>>(x, cls, Ws, bs,
                                          Wqkv + 128 * EN, bqkv + 128,
                                          u, mkeys, attnAcc, oldAcc, Zacc, cntAcc);
    k_final<<<1, 256, 0, stream>>>(sem, Wo, bo, Wr, br, Wg, bg, gamma, beta,
                                   attnAcc, oldAcc, Zacc, cntAcc, fused);
    k_scatter<<<(NNODES * EN / 4) / 256, 256, 0, stream>>>(x, cls, fused, out);
}

extern "C" void kernel_launch(void* const* d_in, const int* in_sizes, int n_in,
                              void* d_out, int out_size, void* d_ws, size_t ws_size,
                              hipStream_t stream)
{
    (void)in_sizes; (void)n_in; (void)out_size; (void)ws_size;
    const float* v       = (const float*)d_in[0];
    const float* c       = (const float*)d_in[1];
    const float* v_sem   = (const float*)d_in[2];
    const float* c_sem   = (const float*)d_in[3];
    const int*   v_class = (const int*)d_in[4];
    const int*   c_class = (const int*)d_in[5];
    const float* Ws_v = (const float*)d_in[6];  const float* bs_v = (const float*)d_in[7];
    const float* Ws_c = (const float*)d_in[8];  const float* bs_c = (const float*)d_in[9];
    const float* Wqkv_v = (const float*)d_in[10]; const float* bqkv_v = (const float*)d_in[11];
    const float* Wo_v = (const float*)d_in[12]; const float* bo_v = (const float*)d_in[13];
    const float* Wqkv_c = (const float*)d_in[14]; const float* bqkv_c = (const float*)d_in[15];
    const float* Wo_c = (const float*)d_in[16]; const float* bo_c = (const float*)d_in[17];
    const float* Wr_v = (const float*)d_in[18]; const float* br_v = (const float*)d_in[19];
    const float* Wr_c = (const float*)d_in[20]; const float* br_c = (const float*)d_in[21];
    const float* Wg_v = (const float*)d_in[22]; const float* bg_v = (const float*)d_in[23];
    const float* Wg_c = (const float*)d_in[24]; const float* bg_c = (const float*)d_in[25];
    const float* gamma = (const float*)d_in[26]; const float* beta = (const float*)d_in[27];

    float* out = (float*)d_out;
    float* wsf = (float*)d_ws;

    run_side(v, v_sem, v_class, Ws_v, bs_v, Wqkv_v, bqkv_v, Wo_v, bo_v,
             Wr_v, br_v, Wg_v, bg_v, gamma, beta, out, wsf, stream);
    run_side(c, c_sem, c_class, Ws_c, bs_c, Wqkv_c, bqkv_c, Wo_c, bo_c,
             Wr_c, br_c, Wg_c, bg_c, gamma, beta,
             out + (size_t)NNODES * EN, wsf + 4096, stream);
}
